// MultiHeadDotProductFastAttention_58188216926814
// MI455X (gfx1250) — matrix operand order repaired and checked by reading, not compiled, not run on hardware
//
#include <hip/hip_runtime.h>

// ---------------------------------------------------------------------------
// MI455X (gfx1250) multi-head attention forward, bf16 WMMA everywhere,
// shared operands staged into LDS via async global->LDS DMA (ASYNCcnt).
// B=2, L=2048, D=1024, H=16, Dh=64
// ---------------------------------------------------------------------------

typedef __attribute__((ext_vector_type(16))) __bf16 v16bf;
typedef __attribute__((ext_vector_type(8)))  __bf16 v8bf;
typedef __attribute__((ext_vector_type(8)))  float  v8f;

#define B_   2
#define L_   2048
#define D_   1024
#define H_   16
#define DH_  64

static __device__ inline v16bf cat8(v8bf lo, v8bf hi) {
  return __builtin_shufflevector(lo, hi, 0,1,2,3,4,5,6,7,8,9,10,11,12,13,14,15);
}

static __device__ inline v8f wmma_bf16(v16bf a, v16bf b, v8f c) {
  // (neg_a, A, neg_b, B, c_mod, C, reuse_a, reuse_b)
  return __builtin_amdgcn_wmma_f32_16x16x32_bf16(false, a, false, b, (short)0, c,
                                                 false, false);
}

// --- CDNA5 async global->LDS copy (per-lane 16B), tracked by ASYNCcnt -------
static __device__ inline unsigned lds_off(const void* p) {
  // generic pointer to LDS: low 32 bits are the wave-relative LDS offset
  return (unsigned)(unsigned long long)(uintptr_t)p;
}
static __device__ inline void async_ld128(unsigned lds, const void* g) {
  asm volatile("global_load_async_to_lds_b128 %0, %1, off"
               :: "v"(lds), "v"((unsigned long long)(uintptr_t)g)
               : "memory");
}
static __device__ inline void wait_async0() {
  asm volatile("s_wait_asynccnt 0x0" ::: "memory");
}

// ---------------------------------------------------------------------------
// Transpose + f32->bf16 convert: W[1024][1024] (row-major) -> Wt[1024][1024]
// with Wt[n][k] = W[k][n].  Makes WMMA B-fragment loads contiguous.
// ---------------------------------------------------------------------------
__global__ __launch_bounds__(256) void k_transpose_bf16(const float* __restrict__ W,
                                                        __bf16* __restrict__ Wt) {
  __shared__ float tile[32][33];
  const int tx = threadIdx.x & 31;
  const int ty = threadIdx.x >> 5;           // 0..7
  const int c0 = blockIdx.x * 32;
  const int r0 = blockIdx.y * 32;
#pragma unroll
  for (int i = 0; i < 4; ++i)
    tile[ty + i * 8][tx] = W[(size_t)(r0 + ty + i * 8) * 1024 + (c0 + tx)];
  __syncthreads();
#pragma unroll
  for (int i = 0; i < 4; ++i)
    Wt[(size_t)(c0 + ty + i * 8) * 1024 + (r0 + tx)] =
        (__bf16)tile[tx][ty + i * 8];
}

// ---------------------------------------------------------------------------
// QKV projection GEMM:  out = act(X[4096x1024] * WtT + bias)
//   mode 0 (Q): scale by 1/sqrt(Dh), store [B,H,L,Dh] bf16
//   mode 1 (K): store [B,H,L,Dh] bf16
//   mode 2 (V): multiply row by mask_k, store transposed [B,H,Dh,L] bf16
// 256 threads = 8 waves; each wave computes a 16x64 strip.  The 64x32 B tile
// (shared by all 8 waves) is staged in LDS via async DMA, double-buffered.
// ---------------------------------------------------------------------------
__global__ __launch_bounds__(256) void k_qkv_gemm(const float* __restrict__ X,
                                                  const __bf16* __restrict__ Wt,
                                                  const float* __restrict__ bias,
                                                  const float* __restrict__ mask,
                                                  __bf16* __restrict__ out,
                                                  int mode) {
  __shared__ __align__(16) __bf16 bbuf[2][64 * 40];  // padded stride 40 -> no bank conflicts
  const int tid  = threadIdx.x;
  const int wave = tid >> 5;
  const int lane = tid & 31;
  const int half = lane >> 4;
  const int l16  = lane & 15;
  const int mTile = blockIdx.x * 8 + wave;     // 16-row tile index (0..255)
  const int nBase = blockIdx.y * 64;

  const v8f vzero = {0.f, 0.f, 0.f, 0.f, 0.f, 0.f, 0.f, 0.f};
  v8f acc[4];
#pragma unroll
  for (int t = 0; t < 4; ++t) acc[t] = vzero;

  const float* arow = X + (size_t)(mTile * 16 + l16) * 1024;

  // stage(buf, kk): 64 rows x 32 k of Wt -> bbuf[buf], one b128 per thread
  const int srow = tid >> 2;        // 0..63
  const int sseg = tid & 3;         // 0..3 (8 bf16 each)
  const __bf16* sgbase = Wt + (size_t)(nBase + srow) * 1024 + sseg * 8;

  async_ld128(lds_off(&bbuf[0][srow * 40 + sseg * 8]), sgbase);
  wait_async0();
  __syncthreads();

  int cur = 0;
  for (int kk = 0; kk < 1024; kk += 32) {
    const int nxt = cur ^ 1;
    if (kk + 32 < 1024) {
      async_ld128(lds_off(&bbuf[nxt][srow * 40 + sseg * 8]), sgbase + kk + 32);
      __builtin_prefetch(arow + kk + 32, 0, 3);
    }

    // Batch all B-fragment LDS loads first (one DScnt drain for 4 WMMAs)
    v16bf Bf[4];
#pragma unroll
    for (int t = 0; t < 4; ++t) {
      const __bf16* bb = &bbuf[cur][(t * 16 + l16) * 40 + half * 8];
      Bf[t] = cat8(*(const v8bf*)bb, *(const v8bf*)(bb + 16));
    }

    // A-fragment (16x32, bf16): per-lane two 8-elem chunks at half*8, 16+half*8
    const float4 a0 = *(const float4*)(arow + kk + half * 8);
    const float4 a1 = *(const float4*)(arow + kk + half * 8 + 4);
    const float4 a2 = *(const float4*)(arow + kk + 16 + half * 8);
    const float4 a3 = *(const float4*)(arow + kk + 16 + half * 8 + 4);
    v16bf A;
    A[0]  = (__bf16)a0.x; A[1]  = (__bf16)a0.y; A[2]  = (__bf16)a0.z; A[3]  = (__bf16)a0.w;
    A[4]  = (__bf16)a1.x; A[5]  = (__bf16)a1.y; A[6]  = (__bf16)a1.z; A[7]  = (__bf16)a1.w;
    A[8]  = (__bf16)a2.x; A[9]  = (__bf16)a2.y; A[10] = (__bf16)a2.z; A[11] = (__bf16)a2.w;
    A[12] = (__bf16)a3.x; A[13] = (__bf16)a3.y; A[14] = (__bf16)a3.z; A[15] = (__bf16)a3.w;

#pragma unroll
    for (int t = 0; t < 4; ++t)
      acc[t] = wmma_bf16(A, Bf[t], acc[t]);

    wait_async0();
    __syncthreads();
    cur = nxt;
  }

  // Epilogue: C/D layout -> lane holds column n; VGPR r holds row (r + 8*half)
#pragma unroll
  for (int t = 0; t < 4; ++t) {
    const int n  = nBase + t * 16 + l16;
    const int h  = n >> 6;
    const int dh = n & 63;
    const float bn = bias[n];
#pragma unroll
    for (int r = 0; r < 8; ++r) {
      const int mrow = mTile * 16 + r + 8 * half;  // flat (b*L + l)
      const int b  = mrow >> 11;
      const int lp = mrow & 2047;
      float c = acc[t][r] + bn;
      size_t idx;
      if (mode == 2) {
        c *= mask[mrow];
        idx = ((size_t)(b * H_ + h) * DH_ + dh) * L_ + lp;   // V transposed
      } else {
        if (mode == 0) c *= 0.125f;                          // 1/sqrt(64)
        idx = ((size_t)(b * H_ + h) * L_ + lp) * DH_ + dh;
      }
      out[idx] = (__bf16)c;
    }
  }
}

// ---------------------------------------------------------------------------
// Flash attention: one block = 64 query rows of one (b,h); 4 waves x 16 rows.
// K/V key-tiles (shared by all 4 waves) staged in LDS via async DMA, double
// buffered; P routed via private padded LDS tile (C/D layout -> A-fragment).
// ---------------------------------------------------------------------------
__global__ __launch_bounds__(128) void k_attn(const __bf16* __restrict__ Q,
                                              const __bf16* __restrict__ K,
                                              const __bf16* __restrict__ Vt,
                                              __bf16* __restrict__ Xo) {
  __shared__ __align__(16) __bf16 pTile[4][16 * 40];  // per-wave, padded
  __shared__ __align__(16) __bf16 kbuf[2][32 * 72];   // [key][dh], stride 72
  __shared__ __align__(16) __bf16 vbuf[2][64 * 40];   // [dh][key], stride 40
  const int tid  = threadIdx.x;
  const int wave = tid >> 5;
  const int lane = tid & 31;
  const int half = lane >> 4;
  const int l16  = lane & 15;
  const int bh    = blockIdx.y;                  // b*H + h
  const int qBase = blockIdx.x * 64 + wave * 16;

  const __bf16* Qb = Q  + (size_t)bh * L_ * DH_;
  const __bf16* Kb = K  + (size_t)bh * L_ * DH_;
  const __bf16* Vb = Vt + (size_t)bh * DH_ * L_;

  // Q A-fragments over Dh=64 (two 16x32 k-chunks), held in registers
  const __bf16* qrow = Qb + (size_t)(qBase + l16) * DH_;
  v16bf Qa[2];
#pragma unroll
  for (int c = 0; c < 2; ++c)
    Qa[c] = cat8(*(const v8bf*)(qrow + c * 32 + half * 8),
                 *(const v8bf*)(qrow + c * 32 + 16 + half * 8));

  const v8f vzero = {0.f, 0.f, 0.f, 0.f, 0.f, 0.f, 0.f, 0.f};
  v8f O[4];
#pragma unroll
  for (int t = 0; t < 4; ++t) O[t] = vzero;
  float mrow[8], lrow[8];
#pragma unroll
  for (int r = 0; r < 8; ++r) { mrow[r] = -__builtin_huge_valf(); lrow[r] = 0.f; }

  __bf16* myP = &pTile[wave][0];

  // K tile: 32 keys x 64 dh = 256 b128-chunks; V tile: 64 dh x 32 keys = 256.
  // 128 threads -> 2 chunks each per tile.
  const int kc0r = tid >> 3,        kc0s = tid & 7;          // K chunk #tid
  const int kc1r = (tid + 128) >> 3, kc1s = (tid + 128) & 7; // K chunk #tid+128
  const int vc0r = tid >> 2,        vc0s = tid & 3;          // V chunk #tid
  const int vc1r = (tid + 128) >> 2, vc1s = (tid + 128) & 3; // V chunk #tid+128

#define STAGE_KV(buf, k0s)                                                      \
  do {                                                                          \
    async_ld128(lds_off(&kbuf[buf][kc0r * 72 + kc0s * 8]),                      \
                Kb + (size_t)((k0s) + kc0r) * DH_ + kc0s * 8);                  \
    async_ld128(lds_off(&kbuf[buf][kc1r * 72 + kc1s * 8]),                      \
                Kb + (size_t)((k0s) + kc1r) * DH_ + kc1s * 8);                  \
    async_ld128(lds_off(&vbuf[buf][vc0r * 40 + vc0s * 8]),                      \
                Vb + (size_t)vc0r * L_ + (k0s) + vc0s * 8);                     \
    async_ld128(lds_off(&vbuf[buf][vc1r * 40 + vc1s * 8]),                      \
                Vb + (size_t)vc1r * L_ + (k0s) + vc1s * 8);                     \
  } while (0)

  STAGE_KV(0, 0);
  wait_async0();
  __syncthreads();

  int cur = 0;
  for (int k0 = 0; k0 < L_; k0 += 32) {
    const int nxt = cur ^ 1;
    if (k0 + 32 < L_) STAGE_KV(nxt, k0 + 32);

    // ---- scores: batch all 4 K-fragment LDS loads, then 4 WMMAs ----
    v16bf Kf[2][2];
#pragma unroll
    for (int c = 0; c < 2; ++c) {
      const __bf16* k0p = &kbuf[cur][l16 * 72 + c * 32 + half * 8];
      const __bf16* k1p = &kbuf[cur][(16 + l16) * 72 + c * 32 + half * 8];
      Kf[0][c] = cat8(*(const v8bf*)k0p, *(const v8bf*)(k0p + 16));
      Kf[1][c] = cat8(*(const v8bf*)k1p, *(const v8bf*)(k1p + 16));
    }
    v8f S0 = vzero, S1 = vzero;
#pragma unroll
    for (int c = 0; c < 2; ++c) {
      S0 = wmma_bf16(Qa[c], Kf[0][c], S0);
      S1 = wmma_bf16(Qa[c], Kf[1][c], S1);
    }

    // ---- online softmax; row reductions across the 16-lane half-groups ----
    float alpha[8];
#pragma unroll
    for (int r = 0; r < 8; ++r) {
      float mx = fmaxf(S0[r], S1[r]);
      mx = fmaxf(mx, __shfl_xor(mx, 1, 32));
      mx = fmaxf(mx, __shfl_xor(mx, 2, 32));
      mx = fmaxf(mx, __shfl_xor(mx, 4, 32));
      mx = fmaxf(mx, __shfl_xor(mx, 8, 32));
      const float mn = fmaxf(mrow[r], mx);
      alpha[r] = __expf(mrow[r] - mn);
      mrow[r] = mn;
      const float p0 = __expf(S0[r] - mn);
      const float p1 = __expf(S1[r] - mn);
      S0[r] = p0; S1[r] = p1;
      float rs = p0 + p1;
      rs += __shfl_xor(rs, 1, 32);
      rs += __shfl_xor(rs, 2, 32);
      rs += __shfl_xor(rs, 4, 32);
      rs += __shfl_xor(rs, 8, 32);
      lrow[r] = lrow[r] * alpha[r] + rs;
    }
#pragma unroll
    for (int t = 0; t < 4; ++t)
#pragma unroll
      for (int r = 0; r < 8; ++r) O[t][r] *= alpha[r];

    // ---- P (C/D layout) -> LDS -> A-fragment layout (per-wave private) ----
#pragma unroll
    for (int r = 0; r < 8; ++r) {
      const int q = r + 8 * half;
      myP[q * 40 + l16]      = (__bf16)S0[r];
      myP[q * 40 + 16 + l16] = (__bf16)S1[r];
    }
    asm volatile("" ::: "memory");  // per-wave LDS: HW keeps DS ops in order
    // Batch Pa + all 4 V-fragment LDS loads, then 4 WMMAs
    v16bf Pa = cat8(*(const v8bf*)(myP + l16 * 40 + half * 8),
                    *(const v8bf*)(myP + l16 * 40 + 16 + half * 8));
    v16bf Vf[4];
#pragma unroll
    for (int t = 0; t < 4; ++t) {
      const __bf16* vp = &vbuf[cur][(t * 16 + l16) * 40 + half * 8];
      Vf[t] = cat8(*(const v8bf*)vp, *(const v8bf*)(vp + 16));
    }
#pragma unroll
    for (int t = 0; t < 4; ++t)
      O[t] = wmma_bf16(Pa, Vf[t], O[t]);

    wait_async0();
    __syncthreads();
    cur = nxt;
  }
#undef STAGE_KV

  // ---- finalize and store X[B,L,H,Dh] bf16 ----
  const int b = bh >> 4;
  const int h = bh & 15;
#pragma unroll
  for (int r = 0; r < 8; ++r) lrow[r] = 1.0f / lrow[r];
#pragma unroll
  for (int t = 0; t < 4; ++t) {
    const int col = h * 64 + t * 16 + l16;
#pragma unroll
    for (int r = 0; r < 8; ++r) {
      const int row = b * L_ + qBase + r + 8 * half;
      Xo[(size_t)row * 1024 + col] = (__bf16)(O[t][r] * lrow[r]);
    }
  }
}

// ---------------------------------------------------------------------------
// Output projection: out[4096x1024] = Xbf[4096x1024] * WoT^T + bo (f32 out),
// B tile staged in LDS via async DMA exactly like k_qkv_gemm.
// ---------------------------------------------------------------------------
__global__ __launch_bounds__(256) void k_out_gemm(const __bf16* __restrict__ Xbf,
                                                  const __bf16* __restrict__ WoT,
                                                  const float* __restrict__ bo,
                                                  float* __restrict__ out) {
  __shared__ __align__(16) __bf16 bbuf[2][64 * 40];
  const int tid  = threadIdx.x;
  const int wave = tid >> 5;
  const int lane = tid & 31;
  const int half = lane >> 4;
  const int l16  = lane & 15;
  const int mTile = blockIdx.x * 8 + wave;
  const int nBase = blockIdx.y * 64;

  const v8f vzero = {0.f, 0.f, 0.f, 0.f, 0.f, 0.f, 0.f, 0.f};
  v8f acc[4];
#pragma unroll
  for (int t = 0; t < 4; ++t) acc[t] = vzero;

  const __bf16* arow = Xbf + (size_t)(mTile * 16 + l16) * 1024;

  const int srow = tid >> 2;
  const int sseg = tid & 3;
  const __bf16* sgbase = WoT + (size_t)(nBase + srow) * 1024 + sseg * 8;

  async_ld128(lds_off(&bbuf[0][srow * 40 + sseg * 8]), sgbase);
  wait_async0();
  __syncthreads();

  int cur = 0;
  for (int kk = 0; kk < 1024; kk += 32) {
    const int nxt = cur ^ 1;
    if (kk + 32 < 1024) {
      async_ld128(lds_off(&bbuf[nxt][srow * 40 + sseg * 8]), sgbase + kk + 32);
      __builtin_prefetch(arow + kk + 32, 0, 3);
    }

    v16bf Bf[4];
#pragma unroll
    for (int t = 0; t < 4; ++t) {
      const __bf16* bb = &bbuf[cur][(t * 16 + l16) * 40 + half * 8];
      Bf[t] = cat8(*(const v8bf*)bb, *(const v8bf*)(bb + 16));
    }
    v16bf A = cat8(*(const v8bf*)(arow + kk + half * 8),
                   *(const v8bf*)(arow + kk + 16 + half * 8));
#pragma unroll
    for (int t = 0; t < 4; ++t)
      acc[t] = wmma_bf16(A, Bf[t], acc[t]);

    wait_async0();
    __syncthreads();
    cur = nxt;
  }

#pragma unroll
  for (int t = 0; t < 4; ++t) {
    const int n = nBase + t * 16 + l16;
    const float bn = bo[n];
#pragma unroll
    for (int r = 0; r < 8; ++r) {
      const int mrow = mTile * 16 + r + 8 * half;
      out[(size_t)mrow * 1024 + n] = acc[t][r] + bn;
    }
  }
}

// ---------------------------------------------------------------------------
extern "C" void kernel_launch(void* const* d_in, const int* in_sizes, int n_in,
                              void* d_out, int out_size, void* d_ws, size_t ws_size,
                              hipStream_t stream) {
  (void)in_sizes; (void)n_in; (void)out_size; (void)ws_size;
  const float* inputs_q  = (const float*)d_in[0];
  const float* inputs_kv = (const float*)d_in[1];
  const float* mask_k    = (const float*)d_in[2];
  const float* Wq = (const float*)d_in[3];
  const float* bq = (const float*)d_in[4];
  const float* Wk = (const float*)d_in[5];
  const float* bk = (const float*)d_in[6];
  const float* Wv = (const float*)d_in[7];
  const float* bv = (const float*)d_in[8];
  const float* Wo = (const float*)d_in[9];
  const float* bo = (const float*)d_in[10];
  float* out = (float*)d_out;

  char* ws = (char*)d_ws;
  const size_t SZ_QKV = (size_t)B_ * H_ * L_ * DH_ * sizeof(__bf16);  // 8 MB
  const size_t SZ_W   = (size_t)1024 * 1024 * sizeof(__bf16);         // 2 MB
  __bf16* Qbf = (__bf16*)(ws + 0 * SZ_QKV);
  __bf16* Kbf = (__bf16*)(ws + 1 * SZ_QKV);
  __bf16* Vt  = (__bf16*)(ws + 2 * SZ_QKV);
  __bf16* Xbf = (__bf16*)(ws + 3 * SZ_QKV);
  __bf16* WqT = (__bf16*)(ws + 4 * SZ_QKV + 0 * SZ_W);
  __bf16* WkT = (__bf16*)(ws + 4 * SZ_QKV + 1 * SZ_W);
  __bf16* WvT = (__bf16*)(ws + 4 * SZ_QKV + 2 * SZ_W);
  __bf16* WoT = (__bf16*)(ws + 4 * SZ_QKV + 3 * SZ_W);

  dim3 tgrid(32, 32);
  k_transpose_bf16<<<tgrid, 256, 0, stream>>>(Wq, WqT);
  k_transpose_bf16<<<tgrid, 256, 0, stream>>>(Wk, WkT);
  k_transpose_bf16<<<tgrid, 256, 0, stream>>>(Wv, WvT);
  k_transpose_bf16<<<tgrid, 256, 0, stream>>>(Wo, WoT);

  dim3 ggrid(32, 16);  // M tiles (4096/128) x N tiles (1024/64)
  k_qkv_gemm<<<ggrid, 256, 0, stream>>>(inputs_q,  WqT, bq, mask_k, Qbf, 0);
  k_qkv_gemm<<<ggrid, 256, 0, stream>>>(inputs_kv, WkT, bk, mask_k, Kbf, 1);
  k_qkv_gemm<<<ggrid, 256, 0, stream>>>(inputs_kv, WvT, bv, mask_k, Vt, 2);

  dim3 agrid(L_ / 64, B_ * H_);
  k_attn<<<agrid, 128, 0, stream>>>(Qbf, Kbf, Vt, Xbf);

  k_out_gemm<<<ggrid, 256, 0, stream>>>(Xbf, WoT, bo, out);
}
